// LongAxisSelfAttention_87522843558319
// MI455X (gfx1250) — matrix operand order repaired and checked by reading, not compiled, not run on hardware
//
#include <hip/hip_runtime.h>

typedef __attribute__((ext_vector_type(16))) _Float16 v16h;
typedef __attribute__((ext_vector_type(8)))  _Float16 v8h;
typedef __attribute__((ext_vector_type(8)))  float    v8f;

#define NUM_HEADS 12
#define SEG_HEADS 6
#define HEAD_DIM  64
#define HDIM      768   /* NUM_HEADS * HEAD_DIM */

#if defined(__has_builtin)
#if __has_builtin(__builtin_amdgcn_tensor_load_to_lds) && __has_builtin(__builtin_amdgcn_s_wait_tensorcnt)
#define USE_TDM 1
#endif
#endif
#ifndef USE_TDM
#define USE_TDM 0
#endif

static __device__ __forceinline__ v8f wmma_f16(v16h a, v16h b, v8f c) {
    // D = A(16x32 f16) * B(32x16 f16) + C(16x16 f32)
    return __builtin_amdgcn_wmma_f32_16x16x32_f16(false, a, false, b, (short)0, c, false, false);
}

#define LDS_OFF(p) ((unsigned)(unsigned long long)(__attribute__((address_space(3))) char*)(p))

#if USE_TDM
typedef unsigned int  tdm_u32x4 __attribute__((ext_vector_type(4)));
typedef int           tdm_i32x8 __attribute__((ext_vector_type(8)));
typedef int           tdm_i32x4 __attribute__((ext_vector_type(4)));

// Issue one Tensor-Data-Mover load of a (tile_d0 x tile_d1 [x tile_d2]) tile of
// 2-byte elements into LDS. stride0 = elements between dim1 lines, stride1 =
// elements between dim2 planes. D# bitfields per CDNA5 ISA 8.3-8.6.
// This toolchain (clang-23 / therock) declares the 6-arg builtin:
//   (u32x4 g0, i32x8 g1, i32x4 g2, i32x4 g3, i32x8 extra, i32 cpol)
static __device__ __forceinline__ void tdm_load_tile(
    unsigned lds_off, unsigned long long gaddr,
    unsigned tile_d0, unsigned tile_d1, unsigned long long stride0,
    unsigned tile_d2, unsigned long long stride1)
{
    tdm_u32x4 g0;
    g0.x = 1u;                                       // count=1, user descriptor
    g0.y = lds_off;                                  // LDS byte address
    g0.z = (unsigned)(gaddr & 0xffffffffull);        // global addr [31:0]
    g0.w = (unsigned)((gaddr >> 32) & 0x01ffffffull) // global addr [56:32]
         | 0x80000000u;                              // type = 2 ("image")

    const unsigned long long td0 = 1ull << 20;       // tensor dims (OOB bounds)
    const unsigned long long td1 = 1ull << 20;
    tdm_i32x8 g1;
    g1[0] = 0x00010000;                              // data_size = 1 (2-byte)
    g1[1] = (int)((td0 & 0xffffull) << 16);          // tensor_dim0[15:0]
    g1[2] = (int)(((td0 >> 16) & 0xffffull) | ((td1 & 0xffffull) << 16));
    g1[3] = (int)(((td1 >> 16) & 0xffffull) | ((unsigned long long)tile_d0 << 16));
    g1[4] = (int)((unsigned long long)tile_d1 | ((unsigned long long)tile_d2 << 16));
    g1[5] = (int)(stride0 & 0xffffffffull);          // tensor_dim0_stride[31:0]
    g1[6] = (int)(((stride0 >> 32) & 0xffffull) | ((stride1 & 0xffffull) << 16));
    g1[7] = (int)((stride1 >> 16) & 0xffffffffull);  // tensor_dim1_stride[47:16]

    tdm_i32x4 g2;
    g2[0] = 1 << 20;                                 // tensor_dim2 bound
    g2[1] = 0; g2[2] = 0; g2[3] = 0;                 // dim3 / iterate: off
    tdm_i32x4 g3 = {0, 0, 0, 0};
    tdm_i32x8 g4 = {0, 0, 0, 0, 0, 0, 0, 0};
    __builtin_amdgcn_tensor_load_to_lds(g0, g1, g2, g3, g4, 0);
}
#endif

// ---------------------------------------------------------------------------
// QKV projection:  C[m,n] = sum_k X[m,k] * W[n,k] + bias[n]
//   transpose_out==0 -> [b][head][s][d] (Q, K); ==1 -> [b][head][d][s] (V)
// 128 threads = 4 waves; 64x64 tile per block; 16x64 strip per wave.
// ---------------------------------------------------------------------------
__global__ __launch_bounds__(128)
void qkv_proj_f16_wmma(const float* __restrict__ X, const float* __restrict__ W,
                       const float* __restrict__ bias, _Float16* __restrict__ out,
                       int transpose_out, int S)
{
    __shared__ __align__(32) _Float16 At[64 * 32];
    __shared__ __align__(32) _Float16 Bt[64 * 32];

    const int tid  = threadIdx.x;
    const int lane = tid & 31;
    const int wave = tid >> 5;
    const int hi   = lane >> 4;
    const int l16  = lane & 15;
    const int mt   = blockIdx.x;
    const int nt   = blockIdx.y;

    v8f acc[4] = {};

    for (int k0 = 0; k0 < HDIM; k0 += 32) {
        for (int i = tid; i < (64 * 32) / 4; i += 128) {
            int row = i >> 3;
            int col = (i & 7) * 4;
            const float4 a = *(const float4*)(X + (size_t)(mt * 64 + row) * HDIM + k0 + col);
            At[row * 32 + col + 0] = (_Float16)a.x;
            At[row * 32 + col + 1] = (_Float16)a.y;
            At[row * 32 + col + 2] = (_Float16)a.z;
            At[row * 32 + col + 3] = (_Float16)a.w;
            const float4 w = *(const float4*)(W + (size_t)(nt * 64 + row) * HDIM + k0 + col);
            Bt[row * 32 + col + 0] = (_Float16)w.x;
            Bt[row * 32 + col + 1] = (_Float16)w.y;
            Bt[row * 32 + col + 2] = (_Float16)w.z;
            Bt[row * 32 + col + 3] = (_Float16)w.w;
        }
        __syncthreads();

        const int arow = wave * 16 + l16;
        v16h afrag = *(const v16h*)(At + arow * 32 + hi * 16);

        #pragma unroll
        for (int jn = 0; jn < 4; ++jn) {
            v16h bfrag = *(const v16h*)(Bt + (jn * 16 + l16) * 32 + hi * 16);
            acc[jn] = wmma_f16(afrag, bfrag, acc[jn]);
        }
        __syncthreads();
    }

    #pragma unroll
    for (int jn = 0; jn < 4; ++jn) {
        int d = jn * 16 + l16;
        float bv = bias[nt * 64 + d];
        #pragma unroll
        for (int i = 0; i < 8; ++i) {
            int m = mt * 64 + wave * 16 + hi * 8 + i;
            int b = m / S, s = m % S;
            float v = acc[jn][i] + bv;
            if (!transpose_out)
                out[(((size_t)b * NUM_HEADS + nt) * S + s) * HEAD_DIM + d] = (_Float16)v;
            else
                out[(((size_t)b * NUM_HEADS + nt) * HEAD_DIM + d) * S + s] = (_Float16)v;
        }
    }
}

// ---------------------------------------------------------------------------
// Flash attention. Grid (S/128, 12, B); 256 threads = 8 waves, 16 q-rows/wave.
// 64-key blocks: per wave 8 QK^T + 8 PV WMMAs per iteration. K/V tiles staged
// by the Tensor Data Mover (one slice per wave) when available.
// ---------------------------------------------------------------------------
__global__ __launch_bounds__(256)
void flash_attn_wmma(const _Float16* __restrict__ Q, const _Float16* __restrict__ K,
                     const _Float16* __restrict__ Vt, float* __restrict__ out, int S)
{
    __shared__ __align__(32) _Float16 Kt[64 * 64];      // [key][d]
    __shared__ __align__(32) _Float16 Vl[64 * 64];      // [d][key]
    __shared__ __align__(32) _Float16 Pl[8 * 16 * 64];  // per-wave P scratch

    const int tid  = threadIdx.x;
    const int lane = tid & 31;
    const int wave = tid >> 5;
    const int hi   = lane >> 4;
    const int l16  = lane & 15;
    const int b = blockIdx.z, h = blockIdx.y, qt = blockIdx.x;

    int stride, s0, qi0, kvlen;
    if (h < SEG_HEADS) {                    // full attention over S keys
        stride = 1; s0 = 0; qi0 = qt * 128; kvlen = S;
    } else {                                // axial: 4 interleaved groups of S/4
        int tpg = S >> 9;                   // (S/4)/128 q-tiles per group
        int g = qt / tpg, t = qt % tpg;
        stride = 4; s0 = g; qi0 = t * 128; kvlen = S >> 2;
    }

    const size_t hbase = ((size_t)b * NUM_HEADS + h) * (size_t)S * HEAD_DIM;
    const _Float16* qh = Q  + hbase;
    const _Float16* kh = K  + hbase;
    const _Float16* vh = Vt + hbase;        // [d][S]

    // Q fragments for this wave's 16 rows (d-chunks 0..31 and 32..63)
    v16h qf[2];
    {
        size_t srow = (size_t)(s0 + stride * (qi0 + wave * 16 + l16)) * HEAD_DIM;
        #pragma unroll
        for (int c = 0; c < 2; ++c) {
            int dc = c * 32;
            qf[c] = *(const v16h*)(qh + srow + dc + hi * 16);
        }
    }

    float m8[8], l8[8];
    v8f accO[4] = {};
    #pragma unroll
    for (int i = 0; i < 8; ++i) { m8[i] = -1e30f; l8[i] = 0.0f; }
    const float scale = 0.125f;             // 1/sqrt(64)

    _Float16* P = Pl + wave * (16 * 64);

    for (int kb = 0; kb < kvlen; kb += 64) {
#if USE_TDM
        // Each wave DMAs one 16-row slice: waves 0-3 -> K tile, 4-7 -> V tile.
        if (wave < 4) {
            unsigned ldso = LDS_OFF(Kt) + (unsigned)wave * 16u * 64u * 2u;
            unsigned long long g = (unsigned long long)(size_t)
                (kh + (size_t)(s0 + stride * (kb + wave * 16)) * HEAD_DIM);
            tdm_load_tile(ldso, g, 64, 16, (unsigned long long)(64 * stride), 0, 0);
        } else {
            int d0 = (wave - 4) * 16;
            unsigned ldso = LDS_OFF(Vl) + (unsigned)d0 * 64u * 2u;
            if (stride == 1) {
                unsigned long long g = (unsigned long long)(size_t)
                    (vh + (size_t)d0 * S + kb);
                tdm_load_tile(ldso, g, 64, 16, (unsigned long long)S, 0, 0);
            } else {          // 3-D: 1-elem lines, 4-elem stride along keys
                unsigned long long g = (unsigned long long)(size_t)
                    (vh + (size_t)d0 * S + s0 + (size_t)4 * kb);
                tdm_load_tile(ldso, g, 1, 64, 4ull, 16, (unsigned long long)S);
            }
        }
        __builtin_amdgcn_s_wait_tensorcnt(0);
#else
        for (int i = tid; i < (64 * 64) / 8; i += 256) {
            int j = i >> 3, col = (i & 7) * 8;
            *(v8h*)(Kt + j * 64 + col) =
                *(const v8h*)(kh + (size_t)(s0 + stride * (kb + j)) * HEAD_DIM + col);
        }
        if (stride == 1) {
            for (int i = tid; i < (64 * 64) / 8; i += 256) {
                int d = i >> 3, col = (i & 7) * 8;
                *(v8h*)(Vl + d * 64 + col) = *(const v8h*)(vh + (size_t)d * S + kb + col);
            }
        } else {
            for (int i = tid; i < 64 * 64; i += 256) {
                int d = i >> 6, j = i & 63;
                Vl[d * 64 + j] = vh[(size_t)d * S + s0 + (size_t)stride * (kb + j)];
            }
        }
#endif
        __syncthreads();

        // ---- scores: sc[n] covers keys kb + n*16 .. kb + n*16+15 ----
        v8f sc[4] = {};
        #pragma unroll
        for (int c = 0; c < 2; ++c) {
            int dc = c * 32;
            #pragma unroll
            for (int n = 0; n < 4; ++n) {
                v16h bf = *(const v16h*)(Kt + (n * 16 + l16) * 64 + dc + hi * 16);
                sc[n] = wmma_f16(qf[c], bf, sc[n]);
            }
        }

        // ---- online softmax over 64 keys ----
        #pragma unroll
        for (int i = 0; i < 8; ++i) {
            float x0 = sc[0][i] * scale, x1 = sc[1][i] * scale;
            float x2 = sc[2][i] * scale, x3 = sc[3][i] * scale;
            float mx = fmaxf(fmaxf(x0, x1), fmaxf(x2, x3));
            #pragma unroll
            for (int off = 1; off < 16; off <<= 1)
                mx = fmaxf(mx, __shfl_xor(mx, off, 32));
            float mnew  = fmaxf(m8[i], mx);
            float alpha = __expf(m8[i] - mnew);
            float p0 = __expf(x0 - mnew), p1 = __expf(x1 - mnew);
            float p2 = __expf(x2 - mnew), p3 = __expf(x3 - mnew);
            float rs = (p0 + p1) + (p2 + p3);
            #pragma unroll
            for (int off = 1; off < 16; off <<= 1)
                rs += __shfl_xor(rs, off, 32);
            l8[i] = l8[i] * alpha + rs;
            m8[i] = mnew;
            #pragma unroll
            for (int j = 0; j < 4; ++j) accO[j][i] *= alpha;
            int row = hi * 8 + i;                 // transpose P into A-layout
            P[row * 64 +      l16] = (_Float16)p0;
            P[row * 64 + 16 + l16] = (_Float16)p1;
            P[row * 64 + 32 + l16] = (_Float16)p2;
            P[row * 64 + 48 + l16] = (_Float16)p3;
        }

        // ---- P fragments (two 16x32 key chunks) and accO += P * V ----
        #pragma unroll
        for (int kc = 0; kc < 2; ++kc) {
            v16h pf = *(const v16h*)(P + l16 * 64 + kc * 32 + hi * 16);
            #pragma unroll
            for (int j = 0; j < 4; ++j) {
                v16h vf = *(const v16h*)(Vl + (j * 16 + l16) * 64 + kc * 32 + hi * 16);
                accO[j] = wmma_f16(pf, vf, accO[j]);
            }
        }
        __syncthreads();
    }

    // ---- finalize: divide by l, store fp32 ctx [b][s][h*64+d] ----
    #pragma unroll
    for (int i = 0; i < 8; ++i) {
        float inv = 1.0f / l8[i];
        int iq = qi0 + wave * 16 + hi * 8 + i;
        int s  = s0 + stride * iq;
        size_t obase = ((size_t)b * S + s) * HDIM + (size_t)h * HEAD_DIM;
        #pragma unroll
        for (int j = 0; j < 4; ++j)
            out[obase + j * 16 + l16] = accO[j][i] * inv;
    }
}

extern "C" void kernel_launch(void* const* d_in, const int* in_sizes, int n_in,
                              void* d_out, int out_size, void* d_ws, size_t ws_size,
                              hipStream_t stream)
{
    const float* X  = (const float*)d_in[0];
    const float* Wq = (const float*)d_in[1];
    const float* bq = (const float*)d_in[2];
    const float* Wk = (const float*)d_in[3];
    const float* bk = (const float*)d_in[4];
    const float* Wv = (const float*)d_in[5];
    const float* bv = (const float*)d_in[6];
    float* out = (float*)d_out;

    const int B = 2;
    const int S = in_sizes[0] / (B * HDIM);   // 4096

    size_t per = (size_t)B * NUM_HEADS * S * HEAD_DIM;  // elements
    _Float16* q16 = (_Float16*)d_ws;
    _Float16* k16 = q16 + per;
    _Float16* v16 = k16 + per;

    dim3 pgrid((B * S) / 64, NUM_HEADS, 1);
    qkv_proj_f16_wmma<<<pgrid, 128, 0, stream>>>(X, Wq, bq, q16, 0, S);
    qkv_proj_f16_wmma<<<pgrid, 128, 0, stream>>>(X, Wk, bk, k16, 0, S);
    qkv_proj_f16_wmma<<<pgrid, 128, 0, stream>>>(X, Wv, bv, v16, 1, S);

    dim3 agrid(S / 128, NUM_HEADS, B);
    flash_attn_wmma<<<agrid, 256, 0, stream>>>(q16, k16, v16, out, S);
}
